// RNN_29222957482282
// MI455X (gfx1250) — matrix operand order repaired and checked by reading, not compiled, not run on hardware
//
#include <hip/hip_runtime.h>
#include <hip/hip_bf16.h>

// ---------------------------------------------------------------------------
// GRU-variant scan for MI455X (gfx1250), bf16 WMMA.
// B=256, T=4096, IN=64, H=128, OUT=64.
// Grid: 16 workgroups (one 16-row batch tile). Block: 256 threads = 8 waves;
// wave w owns gate columns [16w,16w+16) of each r/z/n block.
// All weight B-fragments live in VGPRs (loaded once from global, f32->bf16);
// LDS holds only double-buffered h (16x128) and x_t (16x64) tiles.
// One s_barrier per time step.
// ---------------------------------------------------------------------------

typedef __bf16 bf16;
typedef __attribute__((ext_vector_type(16))) __bf16 v16bf;
typedef __attribute__((ext_vector_type(8)))  __bf16 v8bf;
typedef __attribute__((ext_vector_type(4)))  __bf16 v4bf;
typedef __attribute__((ext_vector_type(8)))  float  v8f;

#define BATCH   256
#define SEQ_T   4096
#define IN_DIM  64
#define H_DIM   128
#define OUT_DIM 64
#define BT      16          // batch tile (WMMA M)

#define SXK 72              // x LDS k-stride (64 + 8 pad)  -> conflict-free
#define SHK 136             // h LDS k-stride (128 + 8 pad) -> conflict-free

union FragU { v16bf v; struct { v8bf lo, hi; } s; };

// A fragment (M=16 x K=32 bf16), row-major LDS with stride `se` elements.
// lanes 0-15: row m=lane, K = kc..kc+7 and kc+16..kc+23
// lanes 16-31: same rows, base kc+8.
__device__ __forceinline__ v16bf ld_a_frag(const bf16* base, int se, int m, int klo) {
    const bf16* p = base + m * se + klo;
    FragU u;
    u.s.lo = *(const v8bf*)(p);
    u.s.hi = *(const v8bf*)(p + 16);
    return u.v;
}

// B fragment from GLOBAL f32 weights: lane holds K = koff..koff+7 and
// koff+16..koff+23 of its column; caller passes &W[n*K] and koff = kc + hi*8.  4x b128 loads.
__device__ __forceinline__ v16bf ld_b_frag_g(const float* rowp, int koff) {
    const float4* p = (const float4*)(rowp + koff);
    float4 a = p[0], b = p[1], c = p[4], d = p[5];
    v16bf r;
    r[0]=(bf16)a.x;  r[1]=(bf16)a.y;  r[2]=(bf16)a.z;  r[3]=(bf16)a.w;
    r[4]=(bf16)b.x;  r[5]=(bf16)b.y;  r[6]=(bf16)b.z;  r[7]=(bf16)b.w;
    r[8]=(bf16)c.x;  r[9]=(bf16)c.y;  r[10]=(bf16)c.z; r[11]=(bf16)c.w;
    r[12]=(bf16)d.x; r[13]=(bf16)d.y; r[14]=(bf16)d.z; r[15]=(bf16)d.w;
    return r;
}

__device__ __forceinline__ v8f wmma_bf16(v16bf a, v16bf b, v8f c) {
    return __builtin_amdgcn_wmma_f32_16x16x32_bf16(false, a, false, b,
                                                   (short)0, c, false, false);
}

__device__ __forceinline__ float fast_rcp(float v) {
    return __builtin_amdgcn_rcpf(v);        // raw v_rcp_f32 (trans pipe)
}
__device__ __forceinline__ float fast_sigmoid(float v) {
    return fast_rcp(1.0f + __expf(-v));
}
__device__ __forceinline__ float fast_tanh(float v) {
    // overflow-safe: tanh(x) = sign(x) * (1 - 2/(exp(2|x|)+1))
    float e = __expf(2.0f * fabsf(v));
    float t = 1.0f - 2.0f * fast_rcp(e + 1.0f);
    return copysignf(t, v);
}
__device__ __forceinline__ float grad_gate(float jf, float tf) {
    // 0.5*(1 - tanh((j - (8+t)*H/T)/0.1))
    return 0.5f * (1.0f - fast_tanh((jf - (8.0f + tf) * 0.03125f) * 10.0f));
}

__global__ __launch_bounds__(256)
void RNN_gru_scan_wmma(const float* __restrict__ x,
                       const float* __restrict__ wih,
                       const float* __restrict__ whh,
                       const float* __restrict__ bias,
                       const float* __restrict__ bias_n,
                       const float* __restrict__ wout,
                       const float* __restrict__ bout,
                       float* __restrict__ out)
{
    __shared__ bf16 sH[2][BT * SHK];   // gated h, double buffered (8704 B)
    __shared__ bf16 sX[2][BT * SXK];   // x_t bf16, double buffered (4608 B)

    const int tid  = threadIdx.x;
    const int lane = tid & 31;
    const int wave = tid >> 5;          // 0..7 -> gate-column tile
    const int ln   = lane & 15;
    const int hi   = lane >> 4;         // 0|1 (half-wave)
    const int jcol = wave * 16 + ln;    // hidden/gate column owned by lane
    const float jf = (float)jcol;

    // ---- per-lane scalar biases (C/D layout: column == lane%16) ----
    const float b_r  = bias[jcol];
    const float b_z  = bias[H_DIM + jcol];
    const float b_gn = bias[2 * H_DIM + jcol];
    const float b_nn = bias_n[jcol];

    // ---- hoist ALL weight B-fragments into VGPRs (loop-invariant) ----
    const int koffI = hi * 8;                      // within 32-wide K chunk
    v16bf BIr[2], BIz[2], BIn[2];                  // gi: kc = 0,32
    #pragma unroll
    for (int k2 = 0; k2 < 2; ++k2) {
        int ko = k2 * 32 + koffI;
        BIr[k2] = ld_b_frag_g(wih + (size_t)(            jcol) * IN_DIM, ko);
        BIz[k2] = ld_b_frag_g(wih + (size_t)(H_DIM     + jcol) * IN_DIM, ko);
        BIn[k2] = ld_b_frag_g(wih + (size_t)(2 * H_DIM + jcol) * IN_DIM, ko);
    }
    v16bf BHr[4], BHz[4], BHn[4];                  // gh: kc = 0,32,64,96
    #pragma unroll
    for (int k4 = 0; k4 < 4; ++k4) {
        int ko = k4 * 32 + koffI;
        BHr[k4] = ld_b_frag_g(whh + (size_t)(            jcol) * H_DIM, ko);
        BHz[k4] = ld_b_frag_g(whh + (size_t)(H_DIM     + jcol) * H_DIM, ko);
        BHn[k4] = ld_b_frag_g(whh + (size_t)(2 * H_DIM + jcol) * H_DIM, ko);
    }

    // ---- x streaming: 256 threads x float4 = 16x64 tile per step ----
    const int xrow = tid >> 4;           // 0..15
    const int xcol = (tid & 15) * 4;     // 0..60
    const float* xp = x + ((size_t)(blockIdx.x * BT + xrow)) * (size_t)SEQ_T * IN_DIM + xcol;

    v8f hreg = {};                       // own columns of h, C/D layout
    float g  = grad_gate(jf, 0.0f);      // gate for t=0
    float4 xr = *(const float4*)(xp);    // x_0

    for (int t = 0; t < SEQ_T; ++t) {
        bf16* hB = sH[t & 1];
        bf16* xB = sX[t & 1];

        // publish x_t (packed ds_store_b64) and gated h
        v4bf xv;
        xv[0] = (bf16)xr.x; xv[1] = (bf16)xr.y;
        xv[2] = (bf16)xr.z; xv[3] = (bf16)xr.w;
        *(v4bf*)(xB + xrow * SXK + xcol) = xv;

        v8f hg;
        #pragma unroll
        for (int r = 0; r < 8; ++r) {
            hg[r] = hreg[r] * g;
            hB[(r + 8 * hi) * SHK + jcol] = (bf16)hg[r];
        }
        __syncthreads();                 // single barrier per step

        // prefetch x_{t+1} and next gate: both hide under the WMMAs
        if (t + 1 < SEQ_T)
            xr = *(const float4*)(xp + (size_t)(t + 1) * IN_DIM);
        float gnext = grad_gate(jf, (float)(t + 1));

        v8f ar = {}, az = {}, agin = {}, aghn = {};
        #pragma unroll
        for (int k2 = 0; k2 < 2; ++k2) {               // gi = x_t @ Wih^T
            v16bf a = ld_a_frag(xB, SXK, ln, k2 * 32 + hi * 8);
            ar   = wmma_bf16(a, BIr[k2], ar);
            az   = wmma_bf16(a, BIz[k2], az);
            agin = wmma_bf16(a, BIn[k2], agin);
        }
        #pragma unroll
        for (int k4 = 0; k4 < 4; ++k4) {               // gh = h @ Whh^T
            v16bf a = ld_a_frag(hB, SHK, ln, k4 * 32 + hi * 8);
            ar   = wmma_bf16(a, BHr[k4], ar);
            az   = wmma_bf16(a, BHz[k4], az);
            aghn = wmma_bf16(a, BHn[k4], aghn);
        }

        #pragma unroll
        for (int r = 0; r < 8; ++r) {
            float rr = fast_sigmoid(ar[r] + b_r);
            float zz = fast_sigmoid(az[r] + b_z);
            float nn = fast_tanh(agin[r] + b_gn + rr * (aghn[r] + b_nn));
            hreg[r] = nn + zz * (hg[r] - nn);
        }
        g = gnext;
    }

    // ---- epilogue: out = sigmoid(h_final @ Wout^T + b_out) ----
    bf16* hB = sH[0];
    #pragma unroll
    for (int r = 0; r < 8; ++r)
        hB[(r + 8 * hi) * SHK + jcol] = (bf16)hreg[r];   // ungated final h
    __syncthreads();

    if (wave < OUT_DIM / 16) {                           // 4 waves of 8
        const int ocol = wave * 16 + ln;
        v8f acc = {};
        #pragma unroll
        for (int k4 = 0; k4 < 4; ++k4) {
            v16bf a  = ld_a_frag(hB, SHK, ln, k4 * 32 + hi * 8);
            v16bf bw = ld_b_frag_g(wout + (size_t)ocol * H_DIM, k4 * 32 + koffI);
            acc = wmma_bf16(a, bw, acc);
        }
        const float bo = bout[ocol];
        #pragma unroll
        for (int r = 0; r < 8; ++r) {
            int brow = blockIdx.x * BT + r + 8 * hi;
            out[(size_t)brow * OUT_DIM + ocol] = fast_sigmoid(acc[r] + bo);
        }
    }
}

extern "C" void kernel_launch(void* const* d_in, const int* in_sizes, int n_in,
                              void* d_out, int out_size, void* d_ws, size_t ws_size,
                              hipStream_t stream) {
    (void)in_sizes; (void)n_in; (void)d_ws; (void)ws_size; (void)out_size;
    const float* x    = (const float*)d_in[0];
    const float* wih  = (const float*)d_in[1];
    const float* whh  = (const float*)d_in[2];
    const float* bias = (const float*)d_in[3];
    const float* bn   = (const float*)d_in[4];
    const float* wo   = (const float*)d_in[5];
    const float* bo   = (const float*)d_in[6];
    float* out        = (float*)d_out;

    RNN_gru_scan_wmma<<<dim3(BATCH / BT), dim3(256), 0, stream>>>(
        x, wih, whh, bias, bn, wo, bo, out);
}
